// DAWNBlock_88124138979392
// MI455X (gfx1250) — matrix operand order repaired and checked by reading, not compiled, not run on hardware
//
#include <hip/hip_runtime.h>

// ---------------------------------------------------------------------------
// DAWN block for MI455X (gfx1250, wave32, WMMA + TDM).
// B=2 S=2048 D=1024 H=16 DH=64 R=256 N=8 KR=128.  BS = B*S = 4096 rows.
// Heavy math: v_wmma_f32_16x16x32_bf16.  GEMM A-tiles staged to LDS by the
// Tensor Data Mover (tensor_load_to_lds, TENSORcnt-synchronized).
// ---------------------------------------------------------------------------

typedef __bf16 bf16_t;
typedef __attribute__((ext_vector_type(16))) __bf16 v16bf;
typedef __attribute__((ext_vector_type(8)))  float  v8f;
typedef __attribute__((ext_vector_type(4)))  unsigned int u32x4;
typedef __attribute__((ext_vector_type(8)))  int          i32x8;
typedef __attribute__((ext_vector_type(4)))  int          i32x4;

union Frag {
    v16bf        v;
    unsigned int u[8];
    uint2        u2[4];
};

#define WMMA_BF16(a, b, c) \
    __builtin_amdgcn_wmma_f32_16x16x32_bf16(false, (a), false, (b), (short)0, (c), false, false)

#if defined(__has_builtin)
#if __has_builtin(__builtin_amdgcn_tensor_load_to_lds) && \
    __has_builtin(__builtin_amdgcn_s_wait_tensorcnt)
#define HAVE_TDM 1
#endif
#endif

// ---------------------------------------------------------------------------
// float32 -> bf16 conversions (weights converted once per launch; cheap)
// ---------------------------------------------------------------------------
__global__ __launch_bounds__(256) void cvt_f32_bf16(const float* __restrict__ in,
                                                    bf16_t* __restrict__ out,
                                                    long long n) {
    long long i = (long long)blockIdx.x * 256 + threadIdx.x;
    long long stride = (long long)gridDim.x * 256;
    for (; i < n; i += stride) out[i] = (bf16_t)in[i];
}

// W_O is used as attn @ W_O^T: store transposed so GEMM B operand is [K][N] row-major.
__global__ __launch_bounds__(256) void cvt_transpose_bf16(const float* __restrict__ in,
                                                          bf16_t* __restrict__ out,
                                                          int Dm) {
    int i = blockIdx.x * 256 + threadIdx.x;
    int r = i / Dm, c = i % Dm;
    out[(long long)c * Dm + r] = (bf16_t)in[(long long)r * Dm + c];
}

// ---------------------------------------------------------------------------
// Row LayerNorm (D=1024), one 256-thread workgroup per row, bf16 output
// ---------------------------------------------------------------------------
__global__ __launch_bounds__(256) void layernorm_rows(const float* __restrict__ x,
                                                      const float* __restrict__ g,
                                                      const float* __restrict__ b,
                                                      bf16_t* __restrict__ out) {
    const int tid = threadIdx.x, lane = tid & 31, wave = tid >> 5;
    const long long row = blockIdx.x;
    const float* xr = x + row * 1024;
    float s = 0.f, q = 0.f;
    #pragma unroll
    for (int i = tid; i < 1024; i += 256) { float v = xr[i]; s += v; q += v * v; }
    #pragma unroll
    for (int off = 16; off; off >>= 1) { s += __shfl_xor(s, off); q += __shfl_xor(q, off); }
    __shared__ float redS[8], redQ[8];
    if (lane == 0) { redS[wave] = s; redQ[wave] = q; }
    __syncthreads();
    float ts = 0.f, tq = 0.f;
    #pragma unroll
    for (int w = 0; w < 8; ++w) { ts += redS[w]; tq += redQ[w]; }
    const float mean = ts * (1.f / 1024.f);
    const float var  = tq * (1.f / 1024.f) - mean * mean;
    const float rstd = rsqrtf(var + 1e-5f);
    bf16_t* orow = out + row * 1024;
    #pragma unroll
    for (int i = tid; i < 1024; i += 256)
        orow[i] = (bf16_t)((xr[i] - mean) * rstd * g[i] + b[i]);
}

// ---------------------------------------------------------------------------
// GEMM tile staging helpers
// ---------------------------------------------------------------------------
// B tile (32 x 128) -> LDS transposed [n][k], stride 36 (TDM can't transpose,
// and the transposed layout makes B fragments contiguous per the ISA layout).
__device__ __forceinline__ void stage_B(const bf16_t* __restrict__ Bz,
                                        __bf16* Bs, int tid, int tileN,
                                        int ldb, int k0) {
    #pragma unroll
    for (int it = 0; it < 2; ++it) {
        int idx = (tid + it * 256) * 8;
        int kk = idx >> 7, n0 = idx & 127;
        uint4 v = *reinterpret_cast<const uint4*>(Bz + (long long)(k0 + kk) * ldb + tileN + n0);
        const __bf16* pv = reinterpret_cast<const __bf16*>(&v);
        #pragma unroll
        for (int i = 0; i < 8; ++i) Bs[(n0 + i) * 36 + kk] = pv[i];
    }
}

#ifdef HAVE_TDM
// A tile (128 rows x 32 k, row stride lda) staged by the Tensor Data Mover.
// D# descriptor: data_size=2B, tile_dim0=32 (contiguous k), tile_dim1=128,
// tensor_dim0_stride=lda.  LDS padding: pad after every 16 dwords (one 64B
// row) by 2 dwords -> padded row stride = 18 dwords = 36 bf16, matching the
// LDS layout used by the WMMA fragment loads.
__device__ __forceinline__ void stage_A_tdm(const bf16_t* __restrict__ A,
                                            __bf16* As, int tileM,
                                            int lda, int k0) {
    const unsigned long long ga =
        (unsigned long long)(A + (long long)tileM * lda + k0);
    const unsigned ldsOff = (unsigned)(unsigned long long)(void*)As;  // addr[31:0] = LDS offset
    u32x4 g0;
    g0.x = 1u;                                      // count=1, valid user descriptor
    g0.y = ldsOff;                                  // lds_addr
    g0.z = (unsigned)ga;                            // global_addr[31:0]
    g0.w = (unsigned)(ga >> 32) | 0x80000000u;      // global_addr[56:32] | type=2
    i32x8 g1;
    g1[0] = (1 << 16)        // data_size = 2 bytes
          | (1 << 20)        // pad_enable
          | (3 << 22)        // pad_interval: 16 dwords
          | (1 << 25);       // pad_amount: 2 dwords
    g1[1] = (32 << 16);      // tensor_dim0 = 32 (lo16 in d1[31:16])
    g1[2] = (128 << 16);     // tensor_dim0 hi = 0, tensor_dim1 = 128
    g1[3] = (32 << 16);      // tensor_dim1 hi = 0, tile_dim0 = 32
    g1[4] = 128;             // tile_dim1 = 128, tile_dim2 = 0
    g1[5] = lda;             // tensor_dim0_stride lo32 (elements)
    g1[6] = 0;
    g1[7] = 0;
    const i32x4 z4 = (i32x4){0, 0, 0, 0};           // <=2D: groups 2/3 unused
    const i32x8 z8 = (i32x8){0, 0, 0, 0, 0, 0, 0, 0};
    // clang-23 / therock-10.0 arity: (g0, g1, g2, g3, g4, cpol)
    __builtin_amdgcn_tensor_load_to_lds(g0, g1, z4, z4, z8, 0);
}
#else
__device__ __forceinline__ void stage_A_manual(const bf16_t* __restrict__ A,
                                               __bf16* As, int tid, int tileM,
                                               int lda, int k0) {
    #pragma unroll
    for (int it = 0; it < 2; ++it) {
        int idx = (tid + it * 256) * 8;
        int m = idx >> 5, kk = idx & 31;
        uint4 v = *reinterpret_cast<const uint4*>(A + (long long)(tileM + m) * lda + k0 + kk);
        uint2* d = reinterpret_cast<uint2*>(As + m * 36 + kk);
        d[0] = make_uint2(v.x, v.y);
        d[1] = make_uint2(v.z, v.w);
    }
}
#endif

// ---------------------------------------------------------------------------
// Generic bf16 WMMA GEMM:  C[M,N](+z-batch) = alpha * A[M,K] @ B[K,N] (+resid)
//  - 128x128 workgroup tile, 8 waves (2M x 4N), each wave 64x32 -> 8 WMMA acc
//  - double-buffered LDS; A tiles via TDM (wave 0 issues, TENSORcnt sync),
//    B tiles via global_load_b128 + transposed ds stores
// ---------------------------------------------------------------------------
__global__ __launch_bounds__(256) void wmma_gemm_bf16(
        const bf16_t* __restrict__ A, const bf16_t* __restrict__ Bm,
        const float* __restrict__ resid, float* __restrict__ outF,
        bf16_t* __restrict__ outB,
        int K, int lda, int ldb, int ldc,
        long long strideB, int strideC, float alpha) {
    __shared__ __bf16 As[2][128 * 36];
    __shared__ __bf16 Bs[2][128 * 36];

    const int tid  = threadIdx.x;
    const int lane = tid & 31;
    const int wave = tid >> 5;
    const int wm   = (wave >> 2) * 64;   // wave M offset 0/64
    const int wn   = (wave & 3) * 32;    // wave N offset 0..96
    const int half = lane >> 4;
    const int l15  = lane & 15;
    const int tileM = blockIdx.y * 128;
    const int tileN = blockIdx.x * 128;
    const int z     = blockIdx.z;
    const bf16_t* Bz = Bm + (long long)z * strideB;

    v8f acc[4][2];
    #pragma unroll
    for (int i = 0; i < 4; ++i)
        #pragma unroll
        for (int j = 0; j < 2; ++j) acc[i][j] = (v8f){0,0,0,0,0,0,0,0};

    const int nk = K >> 5;                 // K / 32

    // prologue: stage tile 0
#ifdef HAVE_TDM
    if (wave == 0) stage_A_tdm(A, As[0], tileM, lda, 0);
#else
    stage_A_manual(A, As[0], tid, tileM, lda, 0);
#endif
    stage_B(Bz, Bs[0], tid, tileN, ldb, 0);
#ifdef HAVE_TDM
    if (wave == 0) __builtin_amdgcn_s_wait_tensorcnt(0);
#endif
    __syncthreads();

    for (int kt = 0; kt < nk; ++kt) {
        const int buf = kt & 1;
        if (kt + 1 < nk) {                  // stage next tile into other buffer
#ifdef HAVE_TDM
            if (wave == 0) stage_A_tdm(A, As[buf ^ 1], tileM, lda, (kt + 1) * 32);
#else
            stage_A_manual(A, As[buf ^ 1], tid, tileM, lda, (kt + 1) * 32);
#endif
            stage_B(Bz, Bs[buf ^ 1], tid, tileN, ldb, (kt + 1) * 32);
        }
        if (kt + 2 < nk) {                  // L2 prefetch hint two tiles ahead
            __builtin_prefetch(A + (long long)(tileM + (tid >> 1)) * lda + (kt + 2) * 32, 0, 1);
            __builtin_prefetch(Bz + (long long)((kt + 2) * 32 + (tid & 31)) * ldb + tileN, 0, 1);
        }

        // --- A fragments: 16x32 bf16, ISA lane/K interleave ---
        Frag af[4];
        const int abase = half * 8;
        #pragma unroll
        for (int ms = 0; ms < 4; ++ms) {
            const int r = wm + ms * 16 + l15;
            const uint2* s0 = reinterpret_cast<const uint2*>(&As[buf][r * 36 + abase]);
            const uint2* s1 = reinterpret_cast<const uint2*>(&As[buf][r * 36 + 16 + abase]);
            af[ms].u2[0] = s0[0]; af[ms].u2[1] = s0[1];
            af[ms].u2[2] = s1[0]; af[ms].u2[3] = s1[1];
        }
        // --- B fragments: 32x16 bf16, lane = column, K contiguous ---
        Frag bfr[2];
        const int kb = half * 8;
        #pragma unroll
        for (int ns = 0; ns < 2; ++ns) {
            const int c = wn + ns * 16 + l15;
            const uint2* sp = reinterpret_cast<const uint2*>(&Bs[buf][c * 36 + kb]);
            bfr[ns].u2[0] = sp[0]; bfr[ns].u2[1] = sp[1];
            bfr[ns].u2[2] = sp[4]; bfr[ns].u2[3] = sp[5];
        }
        #pragma unroll
        for (int ms = 0; ms < 4; ++ms)
            #pragma unroll
            for (int ns = 0; ns < 2; ++ns)
                acc[ms][ns] = WMMA_BF16(af[ms].v, bfr[ns].v, acc[ms][ns]);

#ifdef HAVE_TDM
        if (wave == 0 && kt + 1 < nk) __builtin_amdgcn_s_wait_tensorcnt(0);
#endif
        __syncthreads();
    }

    // --- epilogue: C layout lane l, VGPR j -> row = j + half*8, col = l&15 ---
    const long long cb = (long long)z * strideC;
    #pragma unroll
    for (int ms = 0; ms < 4; ++ms) {
        #pragma unroll
        for (int ns = 0; ns < 2; ++ns) {
            const int row0 = tileM + wm + ms * 16 + half * 8;
            const int col  = tileN + wn + ns * 16 + l15;
            #pragma unroll
            for (int j = 0; j < 8; ++j) {
                const long long idx = (long long)(row0 + j) * ldc + cb + col;
                float v = acc[ms][ns][j] * alpha;
                if (resid) v += resid[idx];
                if (outF) outF[idx] = v;
                if (outB) outB[idx] = (bf16_t)v;
            }
        }
    }
}

// ---------------------------------------------------------------------------
// combine + restore-scale:  h[bs,r] = sum_n wf[bs,n] * p[bs,n,r]
//                           out[bs, n*R + r] = bf16( wr[bs,n] * h[bs,r] )
// ---------------------------------------------------------------------------
__global__ __launch_bounds__(256) void combine_scale(const float* __restrict__ p,
                                                     const float* __restrict__ wf,
                                                     const float* __restrict__ wr,
                                                     bf16_t* __restrict__ out,
                                                     int Rr) {
    const long long i = (long long)blockIdx.x * 256 + threadIdx.x;  // over BS*Rr
    const long long bs = i / Rr;
    const int r = (int)(i % Rr);
    const float* pr  = p + bs * 8 * Rr + r;
    const float* wfr = wf + bs * 8;
    const float* wrr = wr + bs * 8;
    float h = 0.f;
    #pragma unroll
    for (int n = 0; n < 8; ++n) h += wfr[n] * pr[(long long)n * Rr];
    bf16_t* o = out + bs * 8 * Rr + r;
    #pragma unroll
    for (int n = 0; n < 8; ++n) o[(long long)n * Rr] = (bf16_t)(wrr[n] * h);
}

// ---------------------------------------------------------------------------
// Flash attention (causal): Q pre-scaled by 1/sqrt(DH) at the Q GEMM.
// grid = (S/128, B*H); 8 waves/WG, each wave = 16 query rows, DH=64 in regs.
// Online softmax in WMMA C-layout; row reductions via shfl_xor inside the
// 16-lane halves.  P transposed through per-wave LDS into A-fragment layout.
// ---------------------------------------------------------------------------
__global__ __launch_bounds__(256) void flash_attn_causal(const bf16_t* __restrict__ Q,
                                                         const bf16_t* __restrict__ Kb,
                                                         const bf16_t* __restrict__ Vb,
                                                         bf16_t* __restrict__ O) {
    const int tid  = threadIdx.x;
    const int lane = tid & 31;
    const int wave = tid >> 5;
    const int half = lane >> 4;
    const int l15  = lane & 15;
    const int b = blockIdx.y >> 4;
    const int h = blockIdx.y & 15;
    const long long rowBase = (long long)b * 2048;
    const int qtile = blockIdx.x * 128 + wave * 16;
    const int hcol  = h * 64;

    __shared__ __bf16 Kt[32 * 68];          // [key][dh]
    __shared__ __bf16 Vt[64 * 36];          // [dh][key] (transposed)
    __shared__ __bf16 Pt[8][16 * 36];       // per-wave P transpose scratch

    // Q A-fragments (two K=32 chunks across DH=64), loaded once
    Frag qf[2];
    {
        const bf16_t* qp = Q + (rowBase + qtile + l15) * 1024 + hcol;
        #pragma unroll
        for (int f = 0; f < 2; ++f)
            #pragma unroll
            for (int j = 0; j < 8; ++j) {
                const int k2 = f * 32 + ((j >> 2) << 4) + (half << 3) + ((j & 3) << 1);
                qf[f].u[j] = *reinterpret_cast<const unsigned int*>(qp + k2);
            }
    }

    float mprev[8], lsum[8];
    #pragma unroll
    for (int j = 0; j < 8; ++j) { mprev[j] = -3.0e38f; lsum[j] = 0.f; }
    v8f acc[4];
    #pragma unroll
    for (int n = 0; n < 4; ++n) acc[n] = (v8f){0,0,0,0,0,0,0,0};

    const int kmax = blockIdx.x * 128 + 128;        // uniform WG bound (causal)
    for (int k0 = 0; k0 < kmax; k0 += 32) {
        // --- stage K,V tiles (32 keys x 64 dh) ---
        const int idx = tid * 8;
        const int key = idx >> 6, dh0 = idx & 63;
        const long long krow = rowBase + k0 + key;
        const uint4 kv = *reinterpret_cast<const uint4*>(Kb + krow * 1024 + hcol + dh0);
        const uint4 vv = *reinterpret_cast<const uint4*>(Vb + krow * 1024 + hcol + dh0);
        __syncthreads();                            // previous tile reads done
        {
            uint2* d = reinterpret_cast<uint2*>(&Kt[key * 68 + dh0]);
            d[0] = make_uint2(kv.x, kv.y);
            d[1] = make_uint2(kv.z, kv.w);
            const __bf16* pv = reinterpret_cast<const __bf16*>(&vv);
            #pragma unroll
            for (int i = 0; i < 8; ++i) Vt[(dh0 + i) * 36 + key] = pv[i];
        }
        __syncthreads();
        if (k0 >= qtile + 16) continue;             // fully masked for this wave

        // --- scores S = Q @ K^T  (16 x 32) ---
        Frag kf[2][2];
        #pragma unroll
        for (int ns = 0; ns < 2; ++ns) {
            const int c = ns * 16 + l15;            // key column
            #pragma unroll
            for (int f = 0; f < 2; ++f) {
                const uint2* sp = reinterpret_cast<const uint2*>(&Kt[c * 68 + f * 32 + half * 8]);
                kf[ns][f].u2[0] = sp[0]; kf[ns][f].u2[1] = sp[1];
                kf[ns][f].u2[2] = sp[4]; kf[ns][f].u2[3] = sp[5];
            }
        }
        v8f s[2];
        #pragma unroll
        for (int ns = 0; ns < 2; ++ns) {
            v8f zz = (v8f){0,0,0,0,0,0,0,0};
            zz = WMMA_BF16(qf[0].v, kf[ns][0].v, zz);
            s[ns] = WMMA_BF16(qf[1].v, kf[ns][1].v, zz);
        }
        // --- causal mask ---
        #pragma unroll
        for (int ns = 0; ns < 2; ++ns)
            #pragma unroll
            for (int j = 0; j < 8; ++j) {
                const int q = qtile + half * 8 + j;
                const int ky = k0 + ns * 16 + l15;
                if (ky > q) s[ns][j] = -3.0e38f;
            }
        // --- online softmax update ---
        float pr0[8], pr1[8];
        #pragma unroll
        for (int j = 0; j < 8; ++j) {
            float mx = fmaxf(s[0][j], s[1][j]);
            #pragma unroll
            for (int off = 8; off; off >>= 1) mx = fmaxf(mx, __shfl_xor(mx, off));
            const float mnew = fmaxf(mprev[j], mx);
            const float scl = __expf(mprev[j] - mnew);
            pr0[j] = __expf(s[0][j] - mnew);
            pr1[j] = __expf(s[1][j] - mnew);
            float rs = pr0[j] + pr1[j];
            #pragma unroll
            for (int off = 8; off; off >>= 1) rs += __shfl_xor(rs, off);
            lsum[j] = lsum[j] * scl + rs;
            mprev[j] = mnew;
            #pragma unroll
            for (int n = 0; n < 4; ++n) acc[n][j] *= scl;
        }
        // --- transpose P (C-layout -> A-layout) through per-wave LDS ---
        #pragma unroll
        for (int j = 0; j < 8; ++j) {
            Pt[wave][(half * 8 + j) * 36 + l15]      = (bf16_t)pr0[j];
            Pt[wave][(half * 8 + j) * 36 + 16 + l15] = (bf16_t)pr1[j];
        }
        Frag pf;
        {
            const uint2* s0 = reinterpret_cast<const uint2*>(&Pt[wave][l15 * 36 + half * 8]);
            const uint2* s1 = reinterpret_cast<const uint2*>(&Pt[wave][l15 * 36 + 16 + half * 8]);
            pf.u2[0] = s0[0]; pf.u2[1] = s0[1];
            pf.u2[2] = s1[0]; pf.u2[3] = s1[1];
        }
        // --- O += P @ V ---
        #pragma unroll
        for (int n = 0; n < 4; ++n) {
            const int c = n * 16 + l15;             // dh column
            Frag vf;
            const uint2* sv = reinterpret_cast<const uint2*>(&Vt[c * 36 + half * 8]);
            vf.u2[0] = sv[0]; vf.u2[1] = sv[1]; vf.u2[2] = sv[4]; vf.u2[3] = sv[5];
            acc[n] = WMMA_BF16(pf.v, vf.v, acc[n]);
        }
    }

    // --- normalize and store bf16 output [BS, D] ---
    #pragma unroll
    for (int n = 0; n < 4; ++n)
        #pragma unroll
        for (int j = 0; j < 8; ++j) {
            const long long row = rowBase + qtile + half * 8 + j;
            O[row * 1024 + hcol + n * 16 + l15] = (bf16_t)(acc[n][j] / lsum[j]);
        }
}

// ---------------------------------------------------------------------------
// Host orchestration
// ---------------------------------------------------------------------------
extern "C" void kernel_launch(void* const* d_in, const int* in_sizes, int n_in,
                              void* d_out, int out_size, void* d_ws, size_t ws_size,
                              hipStream_t stream) {
    (void)in_sizes; (void)n_in; (void)out_size; (void)ws_size;
    constexpr int BS = 4096, D = 1024, R = 256, Nn = 8, KR = 128;

    const float* x       = (const float*)d_in[0];
    const float* fqk_wQ  = (const float*)d_in[1];
    const float* fqk_wK  = (const float*)d_in[2];
    const float* fv_w    = (const float*)d_in[3];
    const float* rqk_wQ  = (const float*)d_in[4];
    const float* rqk_wK  = (const float*)d_in[5];
    const float* rv_w    = (const float*)d_in[6];
    const float* fkn_w   = (const float*)d_in[7];
    const float* rkn_w   = (const float*)d_in[8];
    const float* f_qk    = (const float*)d_in[9];
    const float* f_v     = (const float*)d_in[10];
    const float* r_qk    = (const float*)d_in[11];
    const float* r_v     = (const float*)d_in[12];
    const float* f_know  = (const float*)d_in[13];
    const float* r_know  = (const float*)d_in[14];
    const float* W_O     = (const float*)d_in[15];
    const float* ln1g    = (const float*)d_in[16];
    const float* ln1b    = (const float*)d_in[17];
    const float* ln2g    = (const float*)d_in[18];
    const float* ln2b    = (const float*)d_in[19];
    float* out = (float*)d_out;

    char* ws = (char*)d_ws;
    size_t off = 0;
    auto alloc = [&](size_t bytes) -> void* {
        void* p = ws + off;
        off += (bytes + 255) & ~size_t(255);
        return p;
    };
    bf16_t* nxb   = (bf16_t*)alloc((size_t)BS * D * 2);
    float*  p     = (float*) alloc((size_t)BS * Nn * R * 4);
    bf16_t* aq    = (bf16_t*)alloc((size_t)BS * Nn * R * 2);
    bf16_t* ak    = (bf16_t*)alloc((size_t)BS * Nn * R * 2);
    bf16_t* av    = (bf16_t*)alloc((size_t)BS * Nn * R * 2);
    bf16_t* Qb    = (bf16_t*)alloc((size_t)BS * D * 2);
    bf16_t* Kbf   = (bf16_t*)alloc((size_t)BS * D * 2);
    bf16_t* Vbf   = (bf16_t*)alloc((size_t)BS * D * 2);
    bf16_t* Ob    = (bf16_t*)alloc((size_t)BS * D * 2);
    float*  x2    = (float*) alloc((size_t)BS * D * 4);
    bf16_t* nx2b  = (bf16_t*)alloc((size_t)BS * D * 2);
    float*  p2    = (float*) alloc((size_t)BS * Nn * KR * 4);
    bf16_t* a2    = (bf16_t*)alloc((size_t)BS * Nn * KR * 2);
    bf16_t* fqk_b = (bf16_t*)alloc((size_t)Nn * D * R * 2);
    bf16_t* fv_b  = (bf16_t*)alloc((size_t)Nn * D * R * 2);
    bf16_t* rqk_b = (bf16_t*)alloc((size_t)Nn * R * D * 2);
    bf16_t* rv_b  = (bf16_t*)alloc((size_t)Nn * R * D * 2);
    bf16_t* fkn_b = (bf16_t*)alloc((size_t)Nn * D * KR * 2);
    bf16_t* rkn_b = (bf16_t*)alloc((size_t)Nn * KR * D * 2);
    bf16_t* wot_b = (bf16_t*)alloc((size_t)D * D * 2);

    // 1) weight conversions
    cvt_f32_bf16<<<4096, 256, 0, stream>>>(f_qk,   fqk_b, (long long)Nn * D * R);
    cvt_f32_bf16<<<4096, 256, 0, stream>>>(f_v,    fv_b,  (long long)Nn * D * R);
    cvt_f32_bf16<<<4096, 256, 0, stream>>>(r_qk,   rqk_b, (long long)Nn * R * D);
    cvt_f32_bf16<<<4096, 256, 0, stream>>>(r_v,    rv_b,  (long long)Nn * R * D);
    cvt_f32_bf16<<<4096, 256, 0, stream>>>(f_know, fkn_b, (long long)Nn * D * KR);
    cvt_f32_bf16<<<4096, 256, 0, stream>>>(r_know, rkn_b, (long long)Nn * KR * D);
    cvt_transpose_bf16<<<(D * D) / 256, 256, 0, stream>>>(W_O, wot_b, D);

    auto GEMM = [&](const bf16_t* A, const bf16_t* B, const float* resid,
                    float* oF, bf16_t* oB, int M, int Ncols, int K,
                    int lda, int ldb, int ldc, long long sB, int sC,
                    float alpha, int batch) {
        dim3 grid(Ncols / 128, M / 128, batch);
        wmma_gemm_bf16<<<grid, 256, 0, stream>>>(A, B, resid, oF, oB,
                                                 K, lda, ldb, ldc, sB, sC, alpha);
    };

    // 2) LN1
    layernorm_rows<<<BS, 256, 0, stream>>>(x, ln1g, ln1b, nxb);

    // 3) p[bs,n,r] = nx @ f_qk[n]   (batched over n)
    GEMM(nxb, fqk_b, nullptr, p, nullptr, BS, R, D, D, R, Nn * R,
         (long long)D * R, R, 1.f, Nn);
    // 4) combine -> scaled restore inputs aq/ak  (bsr -> bs[n*R+r])
    combine_scale<<<(BS * R) / 256, 256, 0, stream>>>(p, fqk_wQ, rqk_wQ, aq, R);
    combine_scale<<<(BS * R) / 256, 256, 0, stream>>>(p, fqk_wK, rqk_wK, ak, R);
    // 5) p[bs,n,r] = nx @ f_v[n]; combine -> av
    GEMM(nxb, fv_b, nullptr, p, nullptr, BS, R, D, D, R, Nn * R,
         (long long)D * R, R, 1.f, Nn);
    combine_scale<<<(BS * R) / 256, 256, 0, stream>>>(p, fv_w, rv_w, av, R);

    // 6) restore GEMMs (K = N*R = 2048, B = r reshaped [N*R, D] contiguous)
    GEMM(aq, rqk_b, nullptr, nullptr, Qb, BS, D, Nn * R, Nn * R, D, D, 0, 0,
         0.125f, 1);                                      // Q * 1/sqrt(DH)
    GEMM(ak, rqk_b, nullptr, nullptr, Kbf, BS, D, Nn * R, Nn * R, D, D, 0, 0, 1.f, 1);
    GEMM(av, rv_b,  nullptr, nullptr, Vbf, BS, D, Nn * R, Nn * R, D, D, 0, 0, 1.f, 1);

    // 7) causal flash attention
    flash_attn_causal<<<dim3(2048 / 128, 2 * 16), 256, 0, stream>>>(Qb, Kbf, Vbf, Ob);

    // 8) x2 = x + attn @ W_O^T
    GEMM(Ob, wot_b, x, x2, nullptr, BS, D, D, D, D, D, 0, 0, 1.f, 1);

    // 9) LN2
    layernorm_rows<<<BS, 256, 0, stream>>>(x2, ln2g, ln2b, nx2b);

    // 10) knowledge path: p2[bs,n,kr] = nx2 @ f_know[n]
    GEMM(nx2b, fkn_b, nullptr, p2, nullptr, BS, KR, D, D, KR, Nn * KR,
         (long long)D * KR, KR, 1.f, Nn);
    combine_scale<<<(BS * KR) / 256, 256, 0, stream>>>(p2, fkn_w, rkn_w, a2, KR);

    // 11) out = x2 + a2 @ r_know.reshape(N*KR, D)
    GEMM(a2, rkn_b, x2, out, nullptr, BS, D, Nn * KR, Nn * KR, D, D, 0, 0, 1.f, 1);
}
